// MultiHeadSelfAttention_31095563223885
// MI455X (gfx1250) — matrix operand order repaired and checked by reading, not compiled, not run on hardware
//
#include <hip/hip_runtime.h>

// Problem constants (from reference)
#define DM 512   // d_model
#define NH 8     // heads
#define HD 64    // head dim
#define SQ 2048  // seq len
#define BB 2     // batch

typedef __attribute__((ext_vector_type(16))) __bf16 bf16x16;
typedef __attribute__((ext_vector_type(8)))  __bf16 bf16x8;
typedef __attribute__((ext_vector_type(8)))  float  f32x8;
typedef __attribute__((ext_vector_type(8)))  unsigned short u16x8;

__device__ __forceinline__ unsigned short f32_to_bf16(float f) {
  unsigned int u = __builtin_bit_cast(unsigned int, f);
  u = (u + 0x7FFFu + ((u >> 16) & 1u)) >> 16;   // round-to-nearest-even
  return (unsigned short)u;
}

__device__ __forceinline__ f32x8 wmma_bf16(bf16x16 a, bf16x16 b, f32x8 c) {
  // v_wmma_f32_16x16x32_bf16  D = A(16x32) * B(32x16) + C
  return __builtin_amdgcn_wmma_f32_16x16x32_bf16(false, a, false, b, (short)0, c,
                                                 false, false);
}

// A-fragment (16x32 bf16, MxK) from row-major matrix p (row stride ld elems).
// Lane l (m = l&15, hi = l>>4) holds K = kBase + {8*hi..8*hi+7, 16+8*hi..16+8*hi+7}.
__device__ __forceinline__ bf16x16 load_a_frag(const unsigned short* p, int ld,
                                               int rowBase, int kBase, int lane) {
  const int m = lane & 15, hi = lane >> 4;
  const unsigned short* r0 = p + (size_t)(rowBase + m) * ld + kBase + 8 * hi;
  bf16x8 c0 = *(const bf16x8*)(r0);
  bf16x8 c1 = *(const bf16x8*)(r0 + 16);
  bf16x16 out;
#pragma unroll
  for (int i = 0; i < 8; ++i) { out[i] = c0[i]; out[i + 8] = c1[i]; }
  return out;
}

// B-fragment (32x16 bf16, KxN) where B[kk][n] = w[nBase+n][kBase+kk]
// (w row-major with row stride ld). Lane l (n = l&15, khi = l>>4) holds
// K = kBase + {8*khi..8*khi+7, 16+8*khi..16+8*khi+7} of row (nBase+n).
__device__ __forceinline__ bf16x16 load_b_frag(const unsigned short* w, int ld,
                                               int nBase, int kBase, int lane) {
  const int n = lane & 15, khi = lane >> 4;
  const unsigned short* r0 = w + (size_t)(nBase + n) * ld + kBase + 8 * khi;
  bf16x8 c0 = *(const bf16x8*)(r0);
  bf16x8 c1 = *(const bf16x8*)(r0 + 16);
  bf16x16 out;
#pragma unroll
  for (int i = 0; i < 8; ++i) { out[i] = c0[i]; out[i + 8] = c1[i]; }
  return out;
}

// ---------------------------------------------------------------- convert
__global__ void cvt_bf16_kernel(const float* __restrict__ in,
                                unsigned short* __restrict__ out, int n) {
  int i = blockIdx.x * blockDim.x + threadIdx.x;
  if (i < n) out[i] = f32_to_bf16(in[i]);
}

// ---------------------------------------------------------------- QKV proj
// grid = (DM/64, (BB*SQ)/128, 3), block = 128 (4 waves).
// Wave owns a 32x64 output tile (2 A frags share 4 B frags -> 8 WMMAs per
// k-step for 12 b128 loads). z selects Q/K/V.
// Q,K stored per-head row-major [bh][s][64]; V stored transposed [bh][64][s].
__global__ void __launch_bounds__(128)
qkv_proj_kernel(const unsigned short* __restrict__ xb,
                const unsigned short* __restrict__ wqb,
                const unsigned short* __restrict__ wkb,
                const unsigned short* __restrict__ wvb,
                const float* __restrict__ bq, const float* __restrict__ bk,
                const float* __restrict__ bv,
                unsigned short* __restrict__ Qh, unsigned short* __restrict__ Kh,
                unsigned short* __restrict__ Vt) {
  const int lane = threadIdx.x & 31;
  const int wave = threadIdx.x >> 5;
  const int mode = blockIdx.z;
  const unsigned short* w = (mode == 0) ? wqb : ((mode == 1) ? wkb : wvb);
  const float* bias = (mode == 0) ? bq : ((mode == 1) ? bk : bv);
  const int rowBase = blockIdx.y * 128 + wave * 32;
  const int nBase = blockIdx.x * 64;

  f32x8 acc[2][4];
#pragma unroll
  for (int g = 0; g < 2; ++g)
#pragma unroll
    for (int t = 0; t < 4; ++t)
#pragma unroll
      for (int r = 0; r < 8; ++r) acc[g][t][r] = 0.0f;

  for (int k = 0; k < DM; k += 32) {
    bf16x16 a0 = load_a_frag(xb, DM, rowBase, k, lane);
    bf16x16 a1 = load_a_frag(xb, DM, rowBase + 16, k, lane);
#pragma unroll
    for (int t = 0; t < 4; ++t) {
      bf16x16 b = load_b_frag(w, DM, nBase + 16 * t, k, lane);
      acc[0][t] = wmma_bf16(a0, b, acc[0][t]);
      acc[1][t] = wmma_bf16(a1, b, acc[1][t]);
    }
  }

  const int hi = lane >> 4;
#pragma unroll
  for (int g = 0; g < 2; ++g) {
    const int rb = rowBase + 16 * g;
#pragma unroll
    for (int t = 0; t < 4; ++t) {
      const int n = nBase + 16 * t + (lane & 15);
      const float bval = bias[n];
      const int h = n >> 6, d = n & 63;
      if (mode < 2) {
        unsigned short* dst = (mode == 0) ? Qh : Kh;
#pragma unroll
        for (int r = 0; r < 8; ++r) {
          const int m = rb + r + 8 * hi;              // global row in [0, B*S)
          const int b = m >> 11, s = m & (SQ - 1);
          dst[(((size_t)(b * NH + h) * SQ) + s) * HD + d] =
              f32_to_bf16(acc[g][t][r] + bval);
        }
      } else {
        // V transposed: rows r are contiguous along s -> one b128 store/lane
        const int m0 = rb + 8 * hi;
        const int b = m0 >> 11, s0 = m0 & (SQ - 1);
        u16x8 pk;
#pragma unroll
        for (int r = 0; r < 8; ++r) pk[r] = f32_to_bf16(acc[g][t][r] + bval);
        *(u16x8*)(Vt + ((size_t)(b * NH + h) * HD + d) * SQ + s0) = pk;
      }
    }
  }
}

// ---------------------------------------------------------------- attention
// grid = (SQ/64, BB*NH), block = 128 (4 waves). Wave owns 16 query rows,
// flash-style online softmax over key tiles of 32.
__global__ void __launch_bounds__(128)
attn_kernel(const unsigned short* __restrict__ Qh,
            const unsigned short* __restrict__ Kh,
            const unsigned short* __restrict__ Vt,
            unsigned short* __restrict__ attnO) {
  const int lane = threadIdx.x & 31;
  const int wave = threadIdx.x >> 5;
  const int bh = blockIdx.y;
  const int qBase = blockIdx.x * 64 + wave * 16;
  const unsigned short* Q = Qh + (size_t)bh * SQ * HD;
  const unsigned short* K = Kh + (size_t)bh * SQ * HD;
  const unsigned short* V = Vt + (size_t)bh * HD * SQ;

  __shared__ __align__(16) unsigned short pbuf[4][16 * 32];  // per-wave P tile
  unsigned short* pb = &pbuf[wave][0];
  const float scale = 0.125f;  // 1/sqrt(64)
  const int hi = lane >> 4;

  bf16x16 aq0 = load_a_frag(Q, HD, qBase, 0, lane);
  bf16x16 aq1 = load_a_frag(Q, HD, qBase, 32, lane);

  f32x8 o[4];
#pragma unroll
  for (int t = 0; t < 4; ++t)
#pragma unroll
    for (int r = 0; r < 8; ++r) o[t][r] = 0.0f;
  float mrow[8], lrow[8];
#pragma unroll
  for (int r = 0; r < 8; ++r) { mrow[r] = -1e30f; lrow[r] = 0.0f; }

  for (int kv = 0; kv < SQ; kv += 32) {
    // scores: 16 queries x 32 keys (two C fragments)
    f32x8 s0, s1;
#pragma unroll
    for (int r = 0; r < 8; ++r) { s0[r] = 0.0f; s1[r] = 0.0f; }
    s0 = wmma_bf16(aq0, load_b_frag(K, HD, kv, 0, lane), s0);
    s0 = wmma_bf16(aq1, load_b_frag(K, HD, kv, 32, lane), s0);
    s1 = wmma_bf16(aq0, load_b_frag(K, HD, kv + 16, 0, lane), s1);
    s1 = wmma_bf16(aq1, load_b_frag(K, HD, kv + 16, 32, lane), s1);

    // online softmax; row m = r + 8*hi lives in one 16-lane group per VGPR r
    float corr[8];
#pragma unroll
    for (int r = 0; r < 8; ++r) {
      float a = s0[r] * scale, b2 = s1[r] * scale;
      float t = fmaxf(a, b2);
      t = fmaxf(t, __shfl_xor(t, 1, 32));
      t = fmaxf(t, __shfl_xor(t, 2, 32));
      t = fmaxf(t, __shfl_xor(t, 4, 32));
      t = fmaxf(t, __shfl_xor(t, 8, 32));
      const float mnew = fmaxf(mrow[r], t);
      corr[r] = __expf(mrow[r] - mnew);
      mrow[r] = mnew;
      const float p0 = __expf(a - mnew), p1 = __expf(b2 - mnew);
      s0[r] = p0; s1[r] = p1;
      float rs = p0 + p1;
      rs += __shfl_xor(rs, 1, 32);
      rs += __shfl_xor(rs, 2, 32);
      rs += __shfl_xor(rs, 4, 32);
      rs += __shfl_xor(rs, 8, 32);
      lrow[r] = lrow[r] * corr[r] + rs;
    }

    // P (C-layout f32) -> LDS row-major bf16 (wave-synchronous, in-order DS)
    {
      const int n = lane & 15;
#pragma unroll
      for (int r = 0; r < 8; ++r) {
        pb[(r + 8 * hi) * 32 + n] = f32_to_bf16(s0[r]);
        pb[(r + 8 * hi) * 32 + n + 16] = f32_to_bf16(s1[r]);
      }
    }
    // reload P as A-fragment (16x32)
    bf16x16 pa;
    {
      const int m = lane & 15;
      const unsigned short* r0 = pb + m * 32 + 8 * hi;
      bf16x8 c0 = *(const bf16x8*)(r0);
      bf16x8 c1 = *(const bf16x8*)(r0 + 16);
#pragma unroll
      for (int i = 0; i < 8; ++i) { pa[i] = c0[i]; pa[i + 8] = c1[i]; }
    }

    // O = O*corr + P x V  (V transposed -> contiguous B-fragment loads)
#pragma unroll
    for (int t = 0; t < 4; ++t) {
#pragma unroll
      for (int r = 0; r < 8; ++r) o[t][r] *= corr[r];
      bf16x16 bv = load_b_frag(V, SQ, 16 * t, kv, lane);
      o[t] = wmma_bf16(pa, bv, o[t]);
    }
  }

  // normalize and store to attnO [b][s][512] as bf16
  const int b = bh >> 3, h = bh & 7;
#pragma unroll
  for (int t = 0; t < 4; ++t) {
    const int d = h * 64 + 16 * t + (lane & 15);
#pragma unroll
    for (int r = 0; r < 8; ++r) {
      const int s = qBase + r + 8 * hi;
      attnO[(size_t)(b * SQ + s) * DM + d] = f32_to_bf16(o[t][r] / lrow[r]);
    }
  }
}

// ---------------------------------------------------------------- out proj
// grid = (DM/64, (BB*SQ)/128), block = 128. Wave owns a 32x64 tile.
__global__ void __launch_bounds__(128)
out_proj_kernel(const unsigned short* __restrict__ aO,
                const unsigned short* __restrict__ wob,
                const float* __restrict__ bo, float* __restrict__ out) {
  const int lane = threadIdx.x & 31;
  const int wave = threadIdx.x >> 5;
  const int rowBase = blockIdx.y * 128 + wave * 32;
  const int nBase = blockIdx.x * 64;

  f32x8 acc[2][4];
#pragma unroll
  for (int g = 0; g < 2; ++g)
#pragma unroll
    for (int t = 0; t < 4; ++t)
#pragma unroll
      for (int r = 0; r < 8; ++r) acc[g][t][r] = 0.0f;

  for (int k = 0; k < DM; k += 32) {
    bf16x16 a0 = load_a_frag(aO, DM, rowBase, k, lane);
    bf16x16 a1 = load_a_frag(aO, DM, rowBase + 16, k, lane);
#pragma unroll
    for (int t = 0; t < 4; ++t) {
      bf16x16 b = load_b_frag(wob, DM, nBase + 16 * t, k, lane);
      acc[0][t] = wmma_bf16(a0, b, acc[0][t]);
      acc[1][t] = wmma_bf16(a1, b, acc[1][t]);
    }
  }

  const int hi = lane >> 4;
#pragma unroll
  for (int g = 0; g < 2; ++g) {
    const int rb = rowBase + 16 * g;
#pragma unroll
    for (int t = 0; t < 4; ++t) {
      const int n = nBase + 16 * t + (lane & 15);
      const float bval = bo[n];
#pragma unroll
      for (int r = 0; r < 8; ++r) {
        const int m = rb + r + 8 * hi;
        out[(size_t)m * DM + n] = acc[g][t][r] + bval;
      }
    }
  }
}

extern "C" void kernel_launch(void* const* d_in, const int* in_sizes, int n_in,
                              void* d_out, int out_size, void* d_ws, size_t ws_size,
                              hipStream_t stream) {
  const float* x  = (const float*)d_in[0];
  const float* wq = (const float*)d_in[1];
  const float* bq = (const float*)d_in[2];
  const float* wk = (const float*)d_in[3];
  const float* bk = (const float*)d_in[4];
  const float* wv = (const float*)d_in[5];
  const float* bv = (const float*)d_in[6];
  const float* wo = (const float*)d_in[7];
  const float* bo = (const float*)d_in[8];
  float* out = (float*)d_out;

  const size_t NX = (size_t)BB * SQ * DM;  // 2,097,152 elems
  const size_t NW = (size_t)DM * DM;       //   262,144 elems
  unsigned short* ws  = (unsigned short*)d_ws;
  unsigned short* xb  = ws;
  unsigned short* wqb = xb + NX;
  unsigned short* wkb = wqb + NW;
  unsigned short* wvb = wkb + NW;
  unsigned short* wob = wvb + NW;
  unsigned short* Qh  = wob + NW;
  unsigned short* Kh  = Qh + NX;
  unsigned short* Vt  = Kh + NX;
  unsigned short* aO  = Vt + NX;  // total: 5*NX + 4*NW elems (~23 MB)

  cvt_bf16_kernel<<<dim3((unsigned)((NX + 255) / 256)), 256, 0, stream>>>(x, xb, (int)NX);
  cvt_bf16_kernel<<<dim3((unsigned)((NW + 255) / 256)), 256, 0, stream>>>(wq, wqb, (int)NW);
  cvt_bf16_kernel<<<dim3((unsigned)((NW + 255) / 256)), 256, 0, stream>>>(wk, wkb, (int)NW);
  cvt_bf16_kernel<<<dim3((unsigned)((NW + 255) / 256)), 256, 0, stream>>>(wv, wvb, (int)NW);
  cvt_bf16_kernel<<<dim3((unsigned)((NW + 255) / 256)), 256, 0, stream>>>(wo, wob, (int)NW);

  dim3 gq(DM / 64, (BB * SQ) / 128, 3);
  qkv_proj_kernel<<<gq, 128, 0, stream>>>(xb, wqb, wkb, wvb, bq, bk, bv, Qh, Kh, Vt);

  dim3 ga(SQ / 64, BB * NH);
  attn_kernel<<<ga, 128, 0, stream>>>(Qh, Kh, Vt, aO);

  dim3 go(DM / 64, (BB * SQ) / 128);
  out_proj_kernel<<<go, 128, 0, stream>>>(aO, wob, bo, out);
}
